// WindowAttention_44143673868292
// MI455X (gfx1250) — matrix operand order repaired and checked by reading, not compiled, not run on hardware
//
#include <hip/hip_runtime.h>
#include <hip/hip_bf16.h>

typedef __attribute__((ext_vector_type(16))) __bf16 v16bf;
typedef __attribute__((ext_vector_type(8)))  __bf16 v8bf;
typedef __attribute__((ext_vector_type(8)))  float  v8f;

#define SCALE_F 0.17677669529663687f  // 32^-0.5
#define NTHR 512
#define NWV  16

__device__ inline v8f wmma_bf16(v16bf a, v16bf b, v8f c) {
    return __builtin_amdgcn_wmma_f32_16x16x32_bf16(
        /*neg_a=*/false, a, /*neg_b=*/false, b,
        /*c_mod=*/(short)0, c, /*reuse_a=*/false, /*reuse_b=*/false);
}

// A fragment: 16x32 bf16 tile from row-major [M][K] storage (ld = K stride).
// ISA layout: per lane-half, elems 0-7 = K hlf*8..+7, elems 8-15 = K 16+hlf*8..+7
//  -> two contiguous 16B runs -> two ds_load_b128.
__device__ inline v16bf ldA(const __bf16* base, int ld, int m0, int kb, int lane) {
    int m = m0 + (lane & 15);
    int hlf = (lane >> 4) & 1;
    const __bf16* p = base + m * ld + kb + hlf * 8;
    v8bf lo = *(const v8bf*)(p);
    v8bf hi = *(const v8bf*)(p + 16);
    return __builtin_shufflevector(lo, hi, 0,1,2,3,4,5,6,7,8,9,10,11,12,13,14,15);
}

// Same A layout, fp32 source (converted): four b128 loads + pack converts.
__device__ inline v16bf ldA32(const float* base, int ld, int m0, int kb, int lane) {
    int m = m0 + (lane & 15);
    int hlf = (lane >> 4) & 1;
    const float* p = base + m * ld + kb + hlf * 8;
    float4 a0 = *(const float4*)(p);
    float4 a1 = *(const float4*)(p + 4);
    float4 b0 = *(const float4*)(p + 16);
    float4 b1 = *(const float4*)(p + 20);
    v16bf a;
    a[0]=(__bf16)a0.x; a[1]=(__bf16)a0.y; a[2]=(__bf16)a0.z; a[3]=(__bf16)a0.w;
    a[4]=(__bf16)a1.x; a[5]=(__bf16)a1.y; a[6]=(__bf16)a1.z; a[7]=(__bf16)a1.w;
    a[8]=(__bf16)b0.x; a[9]=(__bf16)b0.y; a[10]=(__bf16)b0.z; a[11]=(__bf16)b0.w;
    a[12]=(__bf16)b1.x; a[13]=(__bf16)b1.y; a[14]=(__bf16)b1.z; a[15]=(__bf16)b1.w;
    return a;
}

// B fragment: logical B[k][n], storage TRANSPOSED row-major [N][K] (ld = K stride).
// Per lane-half: elems 0-7 = K hlf*8..+7, elems 8-15 = K 16+hlf*8..+7 -> two ds_load_b128.
__device__ inline v16bf ldBt(const __bf16* base, int ld, int n0, int kb, int lane) {
    int n = n0 + (lane & 15);
    int hlf = (lane >> 4) & 1;
    const __bf16* p = base + n * ld + kb + hlf * 8;
    v8bf lo = *(const v8bf*)(p);
    v8bf hi = *(const v8bf*)(p + 16);
    return __builtin_shufflevector(lo, hi, 0,1,2,3,4,5,6,7,8,9,10,11,12,13,14,15);
}

// Relative-position index: pixels p=i>>1 (N=2 repeat), p=(r,c) in 8x8 window.
__device__ inline int relidx(int i, int j) {
    int p1 = i >> 1, p2 = j >> 1;
    int r1 = p1 >> 3, c1 = p1 & 7, r2 = p2 >> 3, c2 = p2 & 7;
    return (r1 - r2 + 7) * 15 + (c1 - c2 + 7);
}

// ---------------------------------------------------------------------------
// Kernel 1: fused per-(window b, head h) attention.
//   grid = (4, 2048), block = 512 (16 waves), dynamic LDS = 303104 B
// LDS map (byte offsets):
//   A    [128][128] bf16   @ 0        (x[b]; later softmax probs p)
//   W    [96][128]  bf16   @ 32768    (qkv weight slice, [N][K]; later tables)
//   tabK/tabQ [256][32] bf16 @ 32768 / 49152   ([m][c] = [N][K] for ldBt)
//   tabVt [32][256] bf16   @ 65536   (v_rpe transposed: [c][m] = [N][K])
//   Q/K  [128][32] bf16    @ 81920 / 90112     ([row][c], serves ldA and ldBt)
//   Vt   [32][128] bf16    @ 98304   ([c][j] = [N][K] for out1 B operand)
//   S1   [128][128] f32    @ 106496  (qk logits)
//   T    [128][256] bf16   @ 172032  (q . k_rpe table)
//   S    [128][256] bf16   @ 237568  (k . q_rpe table)
//   Wacc [128][256] f32    @ 172032  (scatter of p; overlays T+S)
// ---------------------------------------------------------------------------
__global__ __launch_bounds__(NTHR, 1)
void attn_fused_kernel(const float* __restrict__ x,
                       const float* __restrict__ attn_mask,
                       const float* __restrict__ qkv_w,
                       const float* __restrict__ qkv_b,
                       const float* __restrict__ rpe_table,
                       __bf16* __restrict__ x_attn) {
    extern __shared__ __align__(16) char smem[];
    __bf16* A     = (__bf16*)(smem);
    __bf16* W     = (__bf16*)(smem + 32768);
    __bf16* tabK  = (__bf16*)(smem + 32768);
    __bf16* tabQ  = (__bf16*)(smem + 49152);
    __bf16* tabVt = (__bf16*)(smem + 65536);
    __bf16* Q     = (__bf16*)(smem + 81920);
    __bf16* K     = (__bf16*)(smem + 90112);
    __bf16* Vt    = (__bf16*)(smem + 98304);
    float*  S1    = (float*)(smem + 106496);
    __bf16* T     = (__bf16*)(smem + 172032);
    __bf16* Smat  = (__bf16*)(smem + 237568);
    float*  Wacc  = (float*)(smem + 172032);

    const int tid  = threadIdx.x;
    const int lane = tid & 31, wv = tid >> 5;
    const int mlo  = lane & 15, hlf = lane >> 4;
    const int h = blockIdx.x;
    const int b = blockIdx.y;

    // Prefetch this window's mask tile (64 KB) toward L2.
    const float* mbase = attn_mask + (size_t)(b & 127) * 128 * 128;
    __builtin_prefetch(mbase + tid * 32, 0, 1);

    // ---- Stage 0: x[b] -> A (bf16); qkv weight head-slice -> W [96][128] ----
    const float* xb = x + (size_t)b * 128 * 128;
    for (int i = tid; i < 128 * 128; i += NTHR) A[i] = (__bf16)xb[i];
    for (int i = tid; i < 96 * 128; i += NTHR) {
        int n = i >> 7, d = i & 127;                      // n: 0-31 q, 32-63 k, 64-95 v
        int g = (n >> 5) * 128 + h * 32 + (n & 31);       // row of qkv_w [384][128]
        W[i] = (__bf16)qkv_w[g * 128 + d];
    }
    __syncthreads();

    // ---- Stage 1: qkv = x @ w^T + b (q scaled) -> Q/K [i][c], Vt [c][i] ----
    // 16 tiles per segment, one tile per wave; compile-time epilogues.
    {
        const int m0 = (wv & 7) * 16;
        const int nseg = (wv >> 3) * 16;      // 0 or 16 within segment
        const int nl = nseg + mlo;            // 0..31 within segment

        // --- Q segment (scale applied) ---
        v8f acc = {};
#pragma unroll
        for (int ks = 0; ks < 4; ++ks)
            acc = wmma_bf16(ldA(A, 128, m0, ks * 32, lane),
                            ldBt(W, 128, nseg, ks * 32, lane), acc);
        float bias = qkv_b[h * 32 + nl];
#pragma unroll
        for (int r = 0; r < 8; ++r)
            Q[(m0 + r + hlf * 8) * 32 + nl] = (__bf16)((acc[r] + bias) * SCALE_F);

        // --- K segment ---
        acc = (v8f){};
#pragma unroll
        for (int ks = 0; ks < 4; ++ks)
            acc = wmma_bf16(ldA(A, 128, m0, ks * 32, lane),
                            ldBt(W, 128, 32 + nseg, ks * 32, lane), acc);
        bias = qkv_b[128 + h * 32 + nl];
#pragma unroll
        for (int r = 0; r < 8; ++r)
            K[(m0 + r + hlf * 8) * 32 + nl] = (__bf16)(acc[r] + bias);

        // --- V segment (stored transposed, contiguous b128 store) ---
        acc = (v8f){};
#pragma unroll
        for (int ks = 0; ks < 4; ++ks)
            acc = wmma_bf16(ldA(A, 128, m0, ks * 32, lane),
                            ldBt(W, 128, 64 + nseg, ks * 32, lane), acc);
        bias = qkv_b[256 + h * 32 + nl];
        v8bf pk;
#pragma unroll
        for (int r = 0; r < 8; ++r) pk[r] = (__bf16)(acc[r] + bias);
        *(v8bf*)(Vt + nl * 128 + m0 + hlf * 8) = pk;
    }
    __syncthreads();

    // ---- Stage 1b: RPE tables (padded 225->256), q_rpe pre-scaled ----
    for (int i = tid; i < 256 * 32; i += NTHR) {
        int m = i >> 5, c = i & 31;
        float vq = 0.f, vk = 0.f, vv = 0.f;
        if (m < 225) {
            const float* row = rpe_table + (size_t)m * 384 + h * 96;
            vq = row[c] * SCALE_F; vk = row[32 + c]; vv = row[64 + c];
        }
        tabQ[i] = (__bf16)vq; tabK[i] = (__bf16)vk;
        tabVt[c * 256 + m] = (__bf16)vv;
    }
    __syncthreads();

    // ---- Stage 2a: S1 = q_scaled @ k^T (f32) ----
    for (int t = wv; t < 64; t += NWV) {
        int m0 = (t & 7) * 16, n0 = (t >> 3) * 16;
        v8f acc = {};
        acc = wmma_bf16(ldA(Q, 32, m0, 0, lane), ldBt(K, 32, n0, 0, lane), acc);
        int n = n0 + mlo;
#pragma unroll
        for (int r = 0; r < 8; ++r) S1[(m0 + r + hlf * 8) * 128 + n] = acc[r];
    }
    // ---- Stage 2b: T = q_scaled @ tabK^T ; Stage 2c: S = k @ tabQ^T ----
    for (int t = wv; t < 128; t += NWV) {
        int m0 = (t & 7) * 16, n0 = (t >> 3) * 16;
        v8f acc = {};
        acc = wmma_bf16(ldA(Q, 32, m0, 0, lane), ldBt(tabK, 32, n0, 0, lane), acc);
        int n = n0 + mlo;
#pragma unroll
        for (int r = 0; r < 8; ++r) T[(m0 + r + hlf * 8) * 256 + n] = (__bf16)acc[r];
    }
    for (int t = wv; t < 128; t += NWV) {
        int m0 = (t & 7) * 16, n0 = (t >> 3) * 16;
        v8f acc = {};
        acc = wmma_bf16(ldA(K, 32, m0, 0, lane), ldBt(tabQ, 32, n0, 0, lane), acc);
        int n = n0 + mlo;
#pragma unroll
        for (int r = 0; r < 8; ++r) Smat[(m0 + r + hlf * 8) * 256 + n] = (__bf16)acc[r];
    }
    __syncthreads();

    // ---- Stage 3: wave-parallel softmax: one row per wave-iteration ----
    // logit = qk + T[i, relidx] + S[j, relidx] + mask; p (bf16) -> A
    for (int rr = 0; rr < 8; ++rr) {
        int i = wv * 8 + rr;
        const float* mrow = mbase + (size_t)i * 128;
        float l[4];
        float mx = -3.4e38f;
#pragma unroll
        for (int t = 0; t < 4; ++t) {
            int j = lane + 32 * t;
            int m = relidx(i, j);
            float v = S1[i * 128 + j] + (float)T[i * 256 + m]
                      + (float)Smat[j * 256 + m] + mrow[j];
            l[t] = v;
            mx = fmaxf(mx, v);
        }
#pragma unroll
        for (int off = 16; off; off >>= 1) mx = fmaxf(mx, __shfl_xor(mx, off));
        float sum = 0.f;
#pragma unroll
        for (int t = 0; t < 4; ++t) { l[t] = __expf(l[t] - mx); sum += l[t]; }
#pragma unroll
        for (int off = 16; off; off >>= 1) sum += __shfl_xor(sum, off);
        float inv = 1.f / sum;
#pragma unroll
        for (int t = 0; t < 4; ++t)
            A[i * 128 + lane + 32 * t] = (__bf16)(l[t] * inv);
    }
    __syncthreads();

    // ---- Stage 4: w[i,m] = p[i,2*p2] + p[i,2*p2+1] at m=relidx(i,2*p2) ----
    // (the 64 relidx values within a row are distinct -> conflict-free)
    for (int i = tid; i < 128 * 256; i += NTHR) Wacc[i] = 0.f;
    __syncthreads();
    for (int idx = tid; idx < 128 * 64; idx += NTHR) {
        int i = idx >> 6, p2 = idx & 63;
        int m = relidx(i, p2 * 2);
        Wacc[i * 256 + m] = (float)A[i * 128 + 2 * p2] + (float)A[i * 128 + 2 * p2 + 1];
    }
    __syncthreads();

    // ---- Stage 5: out = p @ v + w @ v_rpe -> x_attn bf16 [b][i][h*32+c] ----
    // 16 tiles, one per wave.
    {
        const int m0 = (wv & 7) * 16, n0 = (wv >> 3) * 16;   // n0 in {0,16}
        v8f acc = {};
#pragma unroll
        for (int ks = 0; ks < 4; ++ks)
            acc = wmma_bf16(ldA(A, 128, m0, ks * 32, lane),
                            ldBt(Vt, 128, n0, ks * 32, lane), acc);
#pragma unroll
        for (int ks = 0; ks < 8; ++ks)
            acc = wmma_bf16(ldA32(Wacc, 256, m0, ks * 32, lane),
                            ldBt(tabVt, 256, n0, ks * 32, lane), acc);
        int n = n0 + mlo;
#pragma unroll
        for (int r = 0; r < 8; ++r) {
            int m = m0 + r + hlf * 8;
            x_attn[((size_t)b * 128 + m) * 128 + h * 32 + n] = (__bf16)acc[r];
        }
    }
}

// ---------------------------------------------------------------------------
// Kernel 2: final projection per window: out[b] = x_attn[b] @ proj_w^T + bias
//   grid = 2048, block = 512, dynamic LDS = 64 KB
//   proj_w is [N=128][K=128] row-major already -> direct bf16 copy for ldBt.
// ---------------------------------------------------------------------------
__global__ __launch_bounds__(NTHR, 1)
void proj_kernel(const __bf16* __restrict__ x_attn,
                 const float* __restrict__ proj_w,
                 const float* __restrict__ proj_b,
                 float* __restrict__ out) {
    extern __shared__ __align__(16) char smem[];
    __bf16* A = (__bf16*)smem;              // [128][128]
    __bf16* B = (__bf16*)(smem + 32768);    // [N=128][K=128]

    const int tid = threadIdx.x, b = blockIdx.x;
    const __bf16* src = x_attn + (size_t)b * 16384;
    for (int i = tid; i < 16384; i += NTHR) A[i] = src[i];
    for (int i = tid; i < 16384; i += NTHR) B[i] = (__bf16)proj_w[i];
    __syncthreads();

    const int lane = tid & 31, wv = tid >> 5;
    const int mlo = lane & 15, hlf = lane >> 4;
    for (int t = wv; t < 64; t += NWV) {
        int m0 = (t & 7) * 16, n0 = (t >> 3) * 16;
        v8f acc = {};
#pragma unroll
        for (int ks = 0; ks < 4; ++ks)
            acc = wmma_bf16(ldA(A, 128, m0, ks * 32, lane),
                            ldBt(B, 128, n0, ks * 32, lane), acc);
        int n = n0 + mlo;
        float bias = proj_b[n];
#pragma unroll
        for (int r = 0; r < 8; ++r)
            out[((size_t)b * 128 + (m0 + r + hlf * 8)) * 128 + n] = acc[r] + bias;
    }
}

extern "C" void kernel_launch(void* const* d_in, const int* in_sizes, int n_in,
                              void* d_out, int out_size, void* d_ws, size_t ws_size,
                              hipStream_t stream) {
    const float* x         = (const float*)d_in[0];
    const float* attn_mask = (const float*)d_in[1];
    const float* qkv_w     = (const float*)d_in[2];
    const float* qkv_b     = (const float*)d_in[3];
    const float* rpe_table = (const float*)d_in[4];
    const float* proj_w    = (const float*)d_in[5];
    const float* proj_b    = (const float*)d_in[6];
    float* out = (float*)d_out;

    __bf16* x_attn = (__bf16*)d_ws;  // 2048*128*128 bf16 = 64 MiB scratch

    dim3 grid1(4, 2048);             // (head, window)
    attn_fused_kernel<<<grid1, NTHR, 303104, stream>>>(
        x, attn_mask, qkv_w, qkv_b, rpe_table, x_attn);

    proj_kernel<<<2048, NTHR, 65536, stream>>>(x_attn, proj_w, proj_b, out);
}
